// MOELayer_77601469104292
// MI455X (gfx1250) — compile-verified, matrix operand order checked by reading, not run on hardware
//
#include <hip/hip_runtime.h>

typedef __attribute__((ext_vector_type(16))) __bf16 v16bf;
typedef __attribute__((ext_vector_type(8)))  float  v8f;

#define S_TOK 16384
#define DMODEL 1024
#define NEXP 32
#define HDIM 1024
#define CAP 512

typedef unsigned int  u32;
typedef unsigned short u16;

union Frag { uint4 q[2]; v16bf v; };

__device__ __forceinline__ u16 f2bf(float f) {
    u32 u = __float_as_uint(f);
    u32 r = u + 0x7FFFu + ((u >> 16) & 1u);   // round to nearest even
    return (u16)(r >> 16);
}

// ---------------------------------------------------------------- init
__global__ void init_kernel(u32* __restrict__ disp, float* __restrict__ out,
                            int* __restrict__ inv) {
    int idx = blockIdx.x * blockDim.x + threadIdx.x;
    int stride = gridDim.x * blockDim.x;
    for (int i = idx; i < NEXP * CAP * DMODEL / 2; i += stride) disp[i] = 0u;
    for (int i = idx; i < S_TOK * DMODEL + 1; i += stride) out[i] = 0.0f;
    for (int i = idx; i < NEXP * CAP; i += stride) inv[i] = -1;
}

// ---------------------------------------------------------------- gating
__global__ __launch_bounds__(256) void gating_kernel(
        const float* __restrict__ x, const float* __restrict__ wg,
        float* __restrict__ gatesAll, float* __restrict__ gateArr,
        int* __restrict__ idxArr) {
    int lane = threadIdx.x & 31;
    int wid  = threadIdx.x >> 5;
    int s = blockIdx.x * 8 + wid;
    const float* xs = x + (size_t)s * DMODEL;

    float myLogit = 0.0f;
    for (int e = 0; e < NEXP; ++e) {
        float a = 0.0f;
        const float* we = wg + (size_t)e * DMODEL;
        for (int d = lane; d < DMODEL; d += 32) a += xs[d] * we[d];
        for (int off = 16; off > 0; off >>= 1) a += __shfl_xor(a, off, 32);
        if (lane == e) myLogit = a;
    }
    float mx = myLogit;
    for (int off = 16; off > 0; off >>= 1) mx = fmaxf(mx, __shfl_xor(mx, off, 32));
    float p = __expf(myLogit - mx);
    float sum = p;
    for (int off = 16; off > 0; off >>= 1) sum += __shfl_xor(sum, off, 32);
    float g = p / sum;
    gatesAll[(size_t)s * NEXP + lane] = g;
    float bl = myLogit; int bi = lane;
    for (int off = 16; off > 0; off >>= 1) {
        float ol = __shfl_xor(bl, off, 32);
        int   oi = __shfl_xor(bi, off, 32);
        if (ol > bl || (ol == bl && oi < bi)) { bl = ol; bi = oi; }
    }
    float gbest = __shfl(g, bi, 32);
    if (lane == 0) { idxArr[s] = bi; gateArr[s] = gbest; }
}

// ---------------------------------------------------------------- position / capacity
__global__ __launch_bounds__(1024) void position_kernel(
        const int* __restrict__ idxArr, const float* __restrict__ gateArr,
        int* __restrict__ slotArr, float* __restrict__ gscale,
        int* __restrict__ inv, float* __restrict__ ceArr) {
    __shared__ int wcnt[32 * 32];
    __shared__ int base[NEXP];
    int tid = threadIdx.x;
    int lane = tid & 31, w = tid >> 5;
    if (tid < NEXP) base[tid] = 0;
    __syncthreads();
    for (int chunk = 0; chunk < S_TOK / 1024; ++chunk) {
        int s = chunk * 1024 + tid;
        int e = idxArr[s];
        int cnt = 0, rank = 0;
        for (int j = 0; j < 32; ++j) {
            int ej = __shfl(e, j, 32);
            if (ej == lane) cnt++;
            if (j < lane && ej == e) rank++;
        }
        wcnt[w * 32 + lane] = cnt;
        __syncthreads();
        int pre = 0;
        for (int ww = 0; ww < w; ++ww) pre += wcnt[ww * 32 + e];
        int loc = base[e] + pre + rank;
        bool valid = loc < CAP;
        slotArr[s] = valid ? (e * CAP + loc) : -1;
        gscale[s]  = valid ? gateArr[s] : 0.0f;
        if (valid) inv[e * CAP + loc] = s;
        __syncthreads();
        if (tid < NEXP) {
            int c = 0;
            for (int ww = 0; ww < 32; ++ww) c += wcnt[ww * 32 + tid];
            base[tid] += c;
        }
        __syncthreads();
    }
    if (tid < NEXP) ceArr[tid] = (float)base[tid];
}

// ---------------------------------------------------------------- me reduction
__global__ __launch_bounds__(256) void me_reduce_kernel(
        const float* __restrict__ gatesAll, float* __restrict__ me) {
    __shared__ float red[256];
    int e = blockIdx.x, tid = threadIdx.x;
    float a = 0.0f;
    for (int s = tid; s < S_TOK; s += 256) a += gatesAll[(size_t)s * NEXP + e];
    red[tid] = a; __syncthreads();
    for (int st = 128; st > 0; st >>= 1) {
        if (tid < st) red[tid] += red[tid + st];
        __syncthreads();
    }
    if (tid == 0) me[e] = red[0];
}

__global__ void loss_kernel(const float* __restrict__ me,
                            const float* __restrict__ ce,
                            float* __restrict__ outLoss) {
    int lane = threadIdx.x & 31;
    float v = me[lane] * ce[lane];
    for (int off = 16; off > 0; off >>= 1) v += __shfl_xor(v, off, 32);
    if (lane == 0)
        outLoss[0] = v * ((float)NEXP / ((float)S_TOK * (float)S_TOK));
}

// ---------------------------------------------------------------- dispatch scatter
__global__ __launch_bounds__(256) void dispatch_kernel(
        const float* __restrict__ x, const int* __restrict__ slotArr,
        u16* __restrict__ disp) {
    int s = blockIdx.x;
    int slot = slotArr[s];
    if (slot < 0) return;
    int d = threadIdx.x * 4;
    float4 xv = *reinterpret_cast<const float4*>(x + (size_t)s * DMODEL + d);
    uint2 o;
    o.x = (u32)f2bf(xv.x) | ((u32)f2bf(xv.y) << 16);
    o.y = (u32)f2bf(xv.z) | ((u32)f2bf(xv.w) << 16);
    *reinterpret_cast<uint2*>(disp + (size_t)slot * DMODEL + d) = o;
}

// ---------------------------------------------------------------- weight transpose + bf16
// Wt[e][n][k] = bf16(W[e][k][n]);  K = N = 1024
__global__ __launch_bounds__(256) void transpose_bf16_kernel(
        const float* __restrict__ W, u16* __restrict__ Wt) {
    __shared__ float Ls[32][33];
    int e = blockIdx.z;
    int kb = blockIdx.x * 32;
    int nb = blockIdx.y * 32;
    const float* We = W + (size_t)e * 1024 * 1024;
    u32* Wte = reinterpret_cast<u32*>(Wt + (size_t)e * 1024 * 1024);
    int t = threadIdx.x;
    int k = t >> 3, n4 = (t & 7) * 4;
    float4 v = *reinterpret_cast<const float4*>(We + (size_t)(kb + k) * 1024 + nb + n4);
    Ls[k][n4 + 0] = v.x; Ls[k][n4 + 1] = v.y;
    Ls[k][n4 + 2] = v.z; Ls[k][n4 + 3] = v.w;
    __syncthreads();
#pragma unroll
    for (int i = 0; i < 2; ++i) {
        int q = t + i * 256;
        int n = q >> 4, u = q & 15;
        u32 o = (u32)f2bf(Ls[2 * u][n]) | ((u32)f2bf(Ls[2 * u + 1][n]) << 16);
        Wte[(size_t)(nb + n) * 512 + (kb >> 1) + u] = o;
    }
}

// ---------------------------------------------------------------- expert GEMMs (bf16 WMMA)
// C[512,1024] = A[512,1024](bf16 row-major) @ Bt[1024,1024](bf16, [n][k])
// block tile 128(m) x 256(n), BK=32, double-buffered LDS fed by
// global_load_async_to_lds_b128 (ASYNCcnt pipeline, no VGPR staging)
template <int EPI>
__global__ __launch_bounds__(256) void gemm_kernel(
        const u16* __restrict__ A, const u16* __restrict__ Bt,
        u16* __restrict__ Hout, const int* __restrict__ inv,
        const float* __restrict__ gscale, float* __restrict__ Y) {
    __shared__ __align__(16) u32 As[2][128 * 20];   // [m][K=32] bf16, row stride 40 halves
    __shared__ __align__(16) u32 Bs[2][256 * 20];   // [n][K=32] bf16

    const int e = blockIdx.z;
    const int nBase = blockIdx.x * 256;
    const int mBase = blockIdx.y * 128;
    const u16* Ae = A  + (size_t)e * CAP * 1024;
    const u16* Be = Bt + (size_t)e * 1024 * 1024;

    const int tid = threadIdx.x;
    const int lane = tid & 31;
    const int wid = tid >> 5;
    const int wr = wid >> 2;     // 0..1 (m)
    const int wc = wid & 3;      // 0..3 (n)

    const v8f vzero = {0, 0, 0, 0, 0, 0, 0, 0};
    v8f acc[4][4];
#pragma unroll
    for (int i = 0; i < 4; ++i)
#pragma unroll
        for (int j = 0; j < 4; ++j) acc[i][j] = vzero;

    // per-thread copy slots: u = tid + i*256 -> row = u>>2, 16B segment = u&3
    // global byte offset (row stride 2048B) and LDS uint offset (row stride 20)
    u32 aGOff[2], bGOff[4], aLOff[2], bLOff[4];
#pragma unroll
    for (int i = 0; i < 2; ++i) {
        int u = tid + i * 256;
        aGOff[i] = (u32)((mBase + (u >> 2)) * 2048 + (u & 3) * 16);
        aLOff[i] = (u32)((u >> 2) * 20 + (u & 3) * 4);
    }
#pragma unroll
    for (int i = 0; i < 4; ++i) {
        int u = tid + i * 256;
        bGOff[i] = (u32)((nBase + (u >> 2)) * 2048 + (u & 3) * 16);
        bLOff[i] = (u32)((u >> 2) * 20 + (u & 3) * 4);
    }

    auto issue_tile = [&](int buf, int kt) {
        const u32 k64 = (u32)kt * 64u;   // byte advance along K
#pragma unroll
        for (int i = 0; i < 2; ++i) {
            u32 ldsa = (u32)(size_t)(&As[buf][aLOff[i]]);
            u32 voff = aGOff[i] + k64;
            asm volatile("global_load_async_to_lds_b128 %0, %1, %2"
                         :: "v"(ldsa), "v"(voff), "s"(Ae) : "memory");
        }
#pragma unroll
        for (int i = 0; i < 4; ++i) {
            u32 ldsb = (u32)(size_t)(&Bs[buf][bLOff[i]]);
            u32 voff = bGOff[i] + k64;
            asm volatile("global_load_async_to_lds_b128 %0, %1, %2"
                         :: "v"(ldsb), "v"(voff), "s"(Be) : "memory");
        }
    };

    issue_tile(0, 0);

    const int kb2  = (lane >> 4) * 4;   // K start: lanes 0-15 -> 0, 16-31 -> 8 halves
    const int rsel = lane & 15;

    for (int kt = 0; kt < 32; ++kt) {
        const int cur = kt & 1;
        __syncthreads();                       // buffer cur^1 no longer being read
        if (kt < 31) {
            issue_tile(cur ^ 1, kt + 1);       // prefetch next tile (DMA in flight)
            asm volatile("s_wait_asynccnt 6" ::: "memory");  // tile kt complete (in-order)
        } else {
            asm volatile("s_wait_asynccnt 0" ::: "memory");
        }
        __syncthreads();                       // all waves' writes to cur visible

        Frag bfr[4];
#pragma unroll
        for (int nt = 0; nt < 4; ++nt) {
            int b = (wc * 64 + nt * 16 + rsel) * 20 + kb2;
            bfr[nt].q[0] = *reinterpret_cast<const uint4*>(&Bs[cur][b]);
            bfr[nt].q[1] = *reinterpret_cast<const uint4*>(&Bs[cur][b + 8]);
        }
#pragma unroll
        for (int mt = 0; mt < 4; ++mt) {
            Frag afr;
            int b = (wr * 64 + mt * 16 + rsel) * 20 + kb2;
            afr.q[0] = *reinterpret_cast<const uint4*>(&As[cur][b]);
            afr.q[1] = *reinterpret_cast<const uint4*>(&As[cur][b + 8]);
#pragma unroll
            for (int nt = 0; nt < 4; ++nt)
                acc[mt][nt] = __builtin_amdgcn_wmma_f32_16x16x32_bf16(
                    false, afr.v, false, bfr[nt].v, (short)0, acc[mt][nt],
                    false, false);
        }
    }

    // epilogue
    const int nsel = lane & 15;
    const int msel = (lane >> 4) * 8;
    if (EPI == 0) {
#pragma unroll
        for (int mt = 0; mt < 4; ++mt)
#pragma unroll
            for (int nt = 0; nt < 4; ++nt) {
                int n = nBase + wc * 64 + nt * 16 + nsel;
#pragma unroll
                for (int r = 0; r < 8; ++r) {
                    int m = mBase + wr * 64 + mt * 16 + msel + r;
                    Hout[(size_t)e * CAP * 1024 + (size_t)m * 1024 + n] =
                        f2bf(fmaxf(acc[mt][nt][r], 0.0f));
                }
            }
    } else {
#pragma unroll
        for (int mt = 0; mt < 4; ++mt) {
#pragma unroll
            for (int r = 0; r < 8; ++r) {
                int m = mBase + wr * 64 + mt * 16 + msel + r;
                int t = inv[e * CAP + m];
                if (t < 0) continue;
                float sc = gscale[t];
#pragma unroll
                for (int nt = 0; nt < 4; ++nt) {
                    int n = nBase + wc * 64 + nt * 16 + nsel;
                    Y[(size_t)t * DMODEL + n] = acc[mt][nt][r] * sc;
                }
            }
        }
    }
}

// ---------------------------------------------------------------- launch
extern "C" void kernel_launch(void* const* d_in, const int* in_sizes, int n_in,
                              void* d_out, int out_size, void* d_ws, size_t ws_size,
                              hipStream_t stream) {
    const float* x  = (const float*)d_in[0];
    const float* wg = (const float*)d_in[1];
    const float* w1 = (const float*)d_in[2];
    const float* w2 = (const float*)d_in[3];
    float* out = (float*)d_out;

    char* ws = (char*)d_ws;
    size_t off = 0;
    u16* disp = (u16*)(ws + off);      off += (size_t)NEXP * CAP * DMODEL * 2;   // 32 MB
    u16* hbuf = (u16*)(ws + off);      off += (size_t)NEXP * CAP * HDIM * 2;     // 32 MB
    u16* w1t  = (u16*)(ws + off);      off += (size_t)NEXP * DMODEL * HDIM * 2;  // 64 MB
    u16* w2t  = (u16*)(ws + off);      off += (size_t)NEXP * HDIM * DMODEL * 2;  // 64 MB
    float* gatesAll = (float*)(ws + off); off += (size_t)S_TOK * NEXP * 4;       // 2 MB
    float* gateArr  = (float*)(ws + off); off += (size_t)S_TOK * 4;
    int*   idxArr   = (int*)(ws + off);   off += (size_t)S_TOK * 4;
    float* gscale   = (float*)(ws + off); off += (size_t)S_TOK * 4;
    int*   slotArr  = (int*)(ws + off);   off += (size_t)S_TOK * 4;
    int*   inv      = (int*)(ws + off);   off += (size_t)NEXP * CAP * 4;
    float* me       = (float*)(ws + off); off += 128;
    float* ce       = (float*)(ws + off); off += 128;
    (void)ws_size; (void)in_sizes; (void)n_in; (void)out_size;

    init_kernel<<<2048, 256, 0, stream>>>((u32*)disp, out, inv);
    gating_kernel<<<S_TOK / 8, 256, 0, stream>>>(x, wg, gatesAll, gateArr, idxArr);
    position_kernel<<<1, 1024, 0, stream>>>(idxArr, gateArr, slotArr, gscale, inv, ce);
    me_reduce_kernel<<<NEXP, 256, 0, stream>>>(gatesAll, me);
    loss_kernel<<<1, 32, 0, stream>>>(me, ce, out + (size_t)S_TOK * DMODEL);
    dispatch_kernel<<<S_TOK, 256, 0, stream>>>(x, slotArr, disp);

    dim3 tg(32, 32, 32);
    transpose_bf16_kernel<<<tg, 256, 0, stream>>>(w1, w1t);
    transpose_bf16_kernel<<<tg, 256, 0, stream>>>(w2, w2t);

    dim3 gg(HDIM / 256, CAP / 128, NEXP);
    gemm_kernel<0><<<gg, 256, 0, stream>>>(disp, w1t, hbuf, nullptr, nullptr, nullptr);
    dim3 gg2(DMODEL / 256, CAP / 128, NEXP);
    gemm_kernel<1><<<gg2, 256, 0, stream>>>(hbuf, w2t, nullptr, inv, gscale, out);
}
